// Net_85822036508774
// MI455X (gfx1250) — matrix operand order repaired and checked by reading, not compiled, not run on hardware
//
#include <hip/hip_runtime.h>
#include <hip/hip_bf16.h>

// ---------------------------------------------------------------------------
// Types for CDNA5 WMMA (gfx1250, wave32)
// ---------------------------------------------------------------------------
typedef __attribute__((ext_vector_type(16))) __bf16 v16bf;
typedef __attribute__((ext_vector_type(8)))  float  v8f;

#define NF 32
#define G_GROUPS 64
#define MAX_DEG 10

// ---------------------------------------------------------------------------
// Degree / normalization (computed once, reused by all 21 GCN convs)
// ---------------------------------------------------------------------------
__global__ void degree_cnt_kernel(const int* __restrict__ dstI, int* __restrict__ cnt, int E) {
    int e = blockIdx.x * blockDim.x + threadIdx.x;
    if (e >= E) return;
    atomicAdd(&cnt[dstI[e]], 1);
}

__global__ void norm_kernel(const int* __restrict__ cnt, float* __restrict__ dinv,
                            float* __restrict__ scoef, int n) {
    int i = blockIdx.x * blockDim.x + threadIdx.x;
    if (i >= n) return;
    float deg = (float)cnt[i] + 1.0f;      // self loop
    float di  = rsqrtf(deg);
    dinv[i]  = di;
    scoef[i] = di * di;
}

__global__ void gcnt_kernel(const int* __restrict__ batch, int* __restrict__ gcnt, int n) {
    int i = blockIdx.x * blockDim.x + threadIdx.x;
    if (i >= n) return;
    atomicAdd(&gcnt[batch[i]], 1);
}

// ---------------------------------------------------------------------------
// WMMA GEMM:  OUT[n x 32] = H[n x 32] @ Wsub[32 x 32]
// Wsub(k, j) = W[k * wstride + coloff + j]    (wstride=32 for layers, 128 for policy)
// One wave computes a 16x32 output tile via two v_wmma_f32_16x16x32_bf16.
// Block = 128 threads = 4 waves = 64 rows.
// ---------------------------------------------------------------------------
__global__ void gemm32_wmma(const float* __restrict__ H, const float* __restrict__ W,
                            float* __restrict__ OUT, int n, int wstride, int coloff) {
    __shared__ float Ws[32 * 32];
    // stage weight submatrix into LDS (coalesced)
    for (int i = threadIdx.x; i < 32 * 32; i += blockDim.x) {
        int k = i >> 5, j = i & 31;
        Ws[i] = W[(size_t)k * wstride + coloff + j];
    }
    __syncthreads();

    const int wave  = threadIdx.x >> 5;                 // 0..3
    const int lane  = threadIdx.x & 31;
    const int rowBase = (blockIdx.x * 4 + wave) * 16;
    const int m     = lane & 15;                        // row-in-tile / B column
    const int half  = lane >> 4;                        // 0 or 1
    const int row   = rowBase + m;
    const int lrow  = (row < n) ? row : 0;              // clamp loads; EXEC stays all-1s

    // --- A fragment: 16x32 bf16.  Lane layout (ISA 7.12.2, 16-bit A 16x32):
    // lanes 0-15 hold K in {kb..kb+7, kb+16..kb+23} with kb = half*8.
    const float* hp = H + (size_t)lrow * NF + half * 8;
    v16bf a;
#pragma unroll
    for (int t = 0; t < 8; ++t) a[t]     = (__bf16)hp[t];
#pragma unroll
    for (int t = 0; t < 8; ++t) a[8 + t] = (__bf16)hp[16 + t];

    const int khalf = half * 8;                         // B: lanes16-31 hold K=8..15,24..31
    const bool fullTile = (rowBase + 16 <= n);          // uniform per wave

#pragma unroll
    for (int jt = 0; jt < 2; ++jt) {
        // --- B fragment: 32x16 bf16, element s holds K = khalf+s (s<8), 16+khalf+(s-8) (s>=8); N = m
        v16bf b;
#pragma unroll
        for (int k = 0; k < 16; ++k)
            b[k] = (__bf16)Ws[(khalf + k + (k & 8)) * 32 + jt * 16 + m];

        v8f c = {};
        c = __builtin_amdgcn_wmma_f32_16x16x32_bf16(
                /*neg_a=*/false, a, /*neg_b=*/false, b,
                /*c_mod=*/(short)0, c, /*reuse_a=*/false, /*reuse_b=*/false);

        // --- C store: VGPR v -> row rowBase + v + half*8, col jt*16 + m
        float* op = OUT + (size_t)(rowBase + half * 8) * NF + jt * 16 + m;
        if (fullTile) {
            // fast path: no per-element predication / EXEC save-restore
#pragma unroll
            for (int v = 0; v < 8; ++v) op[(size_t)v * NF] = c[v];
        } else {
#pragma unroll
            for (int v = 0; v < 8; ++v) {
                int r = rowBase + v + half * 8;
                if (r < n) OUT[(size_t)r * NF + jt * 16 + m] = c[v];
            }
        }
    }
}

// ---------------------------------------------------------------------------
// Edge scatter: AGG[dst,:] += HW[src,:] * coef.  8 lanes per edge, float4 per lane.
// Node features (12.8 MB) live in the 192 MB L2; atomics resolve at L2 atomic units.
// ---------------------------------------------------------------------------
__global__ void edge_scatter(const int* __restrict__ srcI, const int* __restrict__ dstI,
                             const float* __restrict__ HW, float* __restrict__ AGG,
                             const float* __restrict__ dinv, long E, int useCoef) {
    long tid = (long)blockIdx.x * blockDim.x + threadIdx.x;
    long e = tid >> 3;
    if (e >= E) return;
    int f = ((int)tid & 7) * 4;
    int s = srcI[e], d = dstI[e];
    float coef = 1.0f;
    if (useCoef) coef = dinv[s] * dinv[d];
    const float4 v = *(const float4*)(HW + (size_t)s * NF + f);
    float* ap = AGG + (size_t)d * NF + f;
    atomicAdd(ap + 0, v.x * coef);
    atomicAdd(ap + 1, v.y * coef);
    atomicAdd(ap + 2, v.z * coef);
    atomicAdd(ap + 3, v.w * coef);
}

// h_out = relu(agg + hw*scoef + b)   (float4 per thread: b128 load/store)
__global__ void finalize_layer(const float* __restrict__ agg, const float* __restrict__ hw,
                               const float* __restrict__ scoef, const float* __restrict__ b,
                               float* __restrict__ hout, int n) {
    long q = (long)blockIdx.x * blockDim.x + threadIdx.x;   // one float4 per thread
    if (q >= (long)n * (NF / 4)) return;
    long t = q * 4;
    int node = (int)(t >> 5), f = (int)(t & 31);
    float  sc = scoef[node];
    float4 av = *(const float4*)(agg + t);
    float4 hv = *(const float4*)(hw + t);
    float4 bv = *(const float4*)(b + f);
    float4 o;
    o.x = fmaxf(av.x + hv.x * sc + bv.x, 0.0f);
    o.y = fmaxf(av.y + hv.y * sc + bv.y, 0.0f);
    o.z = fmaxf(av.z + hv.z * sc + bv.z, 0.0f);
    o.w = fmaxf(av.w + hv.w * sc + bv.w, 0.0f);
    *(float4*)(hout + t) = o;
}

// policy finalize: p = relu(agg + hw*scoef + b); psum[batch[node], coloff+f..] += p
__global__ void finalize_policy(const float* __restrict__ agg, const float* __restrict__ hw,
                                const float* __restrict__ scoef, const float* __restrict__ b,
                                const int* __restrict__ batch, float* __restrict__ psum,
                                int coloff, int n) {
    long q = (long)blockIdx.x * blockDim.x + threadIdx.x;   // one float4 per thread
    if (q >= (long)n * (NF / 4)) return;
    long t = q * 4;
    int node = (int)(t >> 5), f = (int)(t & 31);
    float  sc = scoef[node];
    float4 av = *(const float4*)(agg + t);
    float4 hv = *(const float4*)(hw + t);
    float4 bv = *(const float4*)(b + f);
    float* pp = psum + (size_t)batch[node] * 128 + coloff + f;
    atomicAdd(pp + 0, fmaxf(av.x + hv.x * sc + bv.x, 0.0f));
    atomicAdd(pp + 1, fmaxf(av.y + hv.y * sc + bv.y, 0.0f));
    atomicAdd(pp + 2, fmaxf(av.z + hv.z * sc + bv.z, 0.0f));
    atomicAdd(pp + 3, fmaxf(av.w + hv.w * sc + bv.w, 0.0f));
}

// ---------------------------------------------------------------------------
// MFConv value branch: out = relu(hsum @ Wlin[d] + h @ Wroot[d] + broot[d]);
// segment_max into pool via monotone int-compare atomicMax (valid: values >= 0).
// ---------------------------------------------------------------------------
__global__ void mf_kernel(const float* __restrict__ h, const float* __restrict__ hsum,
                          const int* __restrict__ cnt, const float* __restrict__ Wlin,
                          const float* __restrict__ Wroot, const float* __restrict__ broot,
                          const int* __restrict__ batch, float* __restrict__ pool, int n) {
    long t = (long)blockIdx.x * blockDim.x + threadIdx.x;
    if (t >= (long)n * NF) return;
    int node = (int)(t >> 5), o = (int)(t & 31);
    int d = cnt[node]; if (d > MAX_DEG) d = MAX_DEG;
    const float* wl = Wlin  + (size_t)d * NF * NF;
    const float* wr = Wroot + (size_t)d * NF * NF;
    const float* hs = hsum + (size_t)node * NF;
    const float* hr = h    + (size_t)node * NF;
    float acc = broot[d * NF + o];
#pragma unroll
    for (int i = 0; i < NF; ++i)
        acc += hs[i] * wl[i * NF + o] + hr[i] * wr[i * NF + o];
    acc = fmaxf(acc, 0.0f);
    atomicMax((int*)&pool[(size_t)batch[node] * NF + o], __float_as_int(acc));
}

// value head: relu(pool) -> 20 -> 1, tanh, relu.  One block (32 threads) per group.
__global__ void value_head(const float* __restrict__ pool, const float* __restrict__ Wv1,
                           const float* __restrict__ bv1, const float* __restrict__ Wv2,
                           const float* __restrict__ bv2, float* __restrict__ out) {
    int g = blockIdx.x, l = threadIdx.x;
    __shared__ float pv[32];
    __shared__ float v1[32];
    pv[l] = fmaxf(pool[(size_t)g * NF + l], 0.0f);
    __syncthreads();
    float a = 0.0f;
    if (l < 20) {
        a = bv1[l];
        for (int i = 0; i < NF; ++i) a += pv[i] * Wv1[i * 20 + l];
        a = fmaxf(a, 0.0f);
    }
    v1[l] = a;
    __syncthreads();
    if (l == 0) {
        float s = bv2[0];
        for (int i = 0; i < 20; ++i) s += v1[i] * Wv2[i];
        out[g] = fmaxf(tanhf(s), 0.0f);
    }
}

// policy head: mean-pool -> 128->64 relu -> 64->64 relu.  One block (128 thr) per group.
__global__ void policy_head(const float* __restrict__ psum, const int* __restrict__ gcnt,
                            const float* __restrict__ Wp1, const float* __restrict__ bp1,
                            const float* __restrict__ Wp2, const float* __restrict__ bp2,
                            float* __restrict__ out) {
    int g = blockIdx.x, l = threadIdx.x;
    __shared__ float p[128];
    __shared__ float t1[64];
    float c = fmaxf((float)gcnt[g], 1.0f);
    p[l] = psum[(size_t)g * 128 + l] / c;
    __syncthreads();
    if (l < 64) {
        float a = bp1[l];
        for (int j = 0; j < 128; ++j) a += p[j] * Wp1[j * 64 + l];
        t1[l] = fmaxf(a, 0.0f);
    }
    __syncthreads();
    if (l < 64) {
        float a = bp2[l];
        for (int j = 0; j < 64; ++j) a += t1[j] * Wp2[j * 64 + l];
        out[(size_t)g * 64 + l] = fmaxf(a, 0.0f);
    }
}

// ---------------------------------------------------------------------------
// Orchestration
// ---------------------------------------------------------------------------
extern "C" void kernel_launch(void* const* d_in, const int* in_sizes, int n_in,
                              void* d_out, int out_size, void* d_ws, size_t ws_size,
                              hipStream_t stream) {
    const float* x     = (const float*)d_in[0];
    const int*   ei    = (const int*)d_in[1];
    const int*   batch = (const int*)d_in[2];
    const float* Wg    = (const float*)d_in[3];
    const float* bg    = (const float*)d_in[4];
    const float* Wlin  = (const float*)d_in[5];
    const float* Wroot = (const float*)d_in[6];
    const float* broot = (const float*)d_in[7];
    const float* Wpol  = (const float*)d_in[8];
    const float* bpol  = (const float*)d_in[9];
    const float* Wv1   = (const float*)d_in[10];
    const float* bv1   = (const float*)d_in[11];
    const float* Wv2   = (const float*)d_in[12];
    const float* bv2   = (const float*)d_in[13];
    const float* Wp1   = (const float*)d_in[14];
    const float* bp1   = (const float*)d_in[15];
    const float* Wp2   = (const float*)d_in[16];
    const float* bp2   = (const float*)d_in[17];

    const int  n = in_sizes[0] / NF;
    const long E = in_sizes[1] / 2;
    const int* srcI = ei;
    const int* dstI = ei + E;

    // --- workspace carve (256B-aligned) ---
    size_t off = 0;
    auto carve = [&](size_t bytes) -> void* {
        void* p = (char*)d_ws + off;
        off = (off + bytes + 255) & ~(size_t)255;
        return p;
    };
    float* dinv  = (float*)carve((size_t)n * 4);
    float* scoef = (float*)carve((size_t)n * 4);
    int*   cnt   = (int*)  carve((size_t)n * 4);
    int*   gcnt  = (int*)  carve(G_GROUPS * 4);
    float* hw    = (float*)carve((size_t)n * NF * 4);
    float* agg   = (float*)carve((size_t)n * NF * 4);
    float* h     = (float*)carve((size_t)n * NF * 4);
    float* pool  = (float*)carve(G_GROUPS * NF * 4);
    float* psum  = (float*)carve(G_GROUPS * 128 * 4);

    const int TB = 256;
    const int gridE1   = (int)((E + TB - 1) / TB);            // 1 thread / edge
    const int gridE8   = (int)((E * 8 + TB - 1) / TB);        // 8 threads / edge
    const int gridN    = (n + TB - 1) / TB;
    const long nelem   = (long)n * NF;
    const int gridNE   = (int)((nelem + TB - 1) / TB);        // 1 float / thread
    const int gridNE4  = (int)((nelem / 4 + TB - 1) / TB);    // 1 float4 / thread
    const int gridGemm = (n + 63) / 64;                       // 64 rows / block

    // degree + norms + group counts (once)
    hipMemsetAsync(cnt,  0, (size_t)n * 4, stream);
    hipMemsetAsync(gcnt, 0, G_GROUPS * 4, stream);
    hipMemsetAsync(pool, 0, G_GROUPS * NF * 4, stream);
    hipMemsetAsync(psum, 0, G_GROUPS * 128 * 4, stream);
    degree_cnt_kernel<<<gridE1, TB, 0, stream>>>(dstI, cnt, (int)E);
    norm_kernel<<<gridN, TB, 0, stream>>>(cnt, dinv, scoef, n);
    gcnt_kernel<<<gridN, TB, 0, stream>>>(batch, gcnt, n);

    // 20 stacked GCNConv layers
    const float* hin = x;
    for (int L = 0; L < 20; ++L) {
        gemm32_wmma<<<gridGemm, 128, 0, stream>>>(hin, Wg + (size_t)L * NF * NF, hw, n, NF, 0);
        hipMemsetAsync(agg, 0, (size_t)n * NF * 4, stream);
        edge_scatter<<<gridE8, TB, 0, stream>>>(srcI, dstI, hw, agg, dinv, E, 1);
        finalize_layer<<<gridNE4, TB, 0, stream>>>(agg, hw, scoef, bg + (size_t)L * NF, h, n);
        hin = h;
    }

    // value branch: neighbor sum (coef=1) into agg, then per-degree MFConv + max-pool
    hipMemsetAsync(agg, 0, (size_t)n * NF * 4, stream);
    edge_scatter<<<gridE8, TB, 0, stream>>>(srcI, dstI, h, agg, dinv, E, 0);
    mf_kernel<<<gridNE, TB, 0, stream>>>(h, agg, cnt, Wlin, Wroot, broot, batch, pool, n);
    value_head<<<G_GROUPS, 32, 0, stream>>>(pool, Wv1, bv1, Wv2, bv2,
                                            (float*)d_out + G_GROUPS * 64);

    // policy branch: GCNConv 32->128 in four 32-col chunks (reuses hw/agg scratch)
    for (int c = 0; c < 4; ++c) {
        gemm32_wmma<<<gridGemm, 128, 0, stream>>>(h, Wpol, hw, n, 128, c * NF);
        hipMemsetAsync(agg, 0, (size_t)n * NF * 4, stream);
        edge_scatter<<<gridE8, TB, 0, stream>>>(srcI, dstI, hw, agg, dinv, E, 1);
        finalize_policy<<<gridNE4, TB, 0, stream>>>(agg, hw, scoef, bpol + (size_t)c * NF,
                                                    batch, psum, c * NF, n);
    }
    policy_head<<<G_GROUPS, 128, 0, stream>>>(psum, gcnt, Wp1, bp1, Wp2, bp2, (float*)d_out);
}
